// SelfAttention_v2_68393059222243
// MI455X (gfx1250) — matrix operand order repaired and checked by reading, not compiled, not run on hardware
//
#include <hip/hip_runtime.h>
#include <hip/hip_bf16.h>

// ---------------------------------------------------------------------------
// Types for CDNA5 WMMA (wave32, 16x16x32 bf16 -> f32)
// ---------------------------------------------------------------------------
typedef __bf16 bf16_t;
typedef __attribute__((ext_vector_type(16))) __bf16 v16bf;
typedef __attribute__((ext_vector_type(8)))  __bf16 v8bf;
typedef __attribute__((ext_vector_type(8)))  float  v8f;

#define SEQ_T 4096
#define DIM_D 1024

// ---------------------------------------------------------------------------
// Fragment loaders per CDNA5 ISA 7.12.2 VGPR layouts (wave32).
//
// A (16x32, bf16), row-major source with leading dim ld:
//   lane L: row = L&15, koff = (L>>4)*8
//   v16bf elements 0..7  = K = koff+0..7        (VGPRs 0..3)
//   v16bf elements 8..15 = K = koff+16..23      (VGPRs 4..7)
// Two contiguous 8xbf16 (16B) loads per lane.
// ---------------------------------------------------------------------------
__device__ __forceinline__ v16bf load_frag_a(const bf16_t* __restrict__ p, int ld) {
  const int lane = threadIdx.x & 31;
  const int r    = lane & 15;
  const int koff = (lane >> 4) << 3;
  const bf16_t* b = p + r * ld + koff;
  v8bf lo = *(const v8bf*)(b);
  v8bf hi = *(const v8bf*)(b + 16);
  v16bf a;
#pragma unroll
  for (int e = 0; e < 8; ++e) { a[e] = lo[e]; a[e + 8] = hi[e]; }
  return a;
}

// B (32x16, bf16) where B[k,n] = W[n0+n, k] with W row-major, leading dim ld:
//   lane L: n = L&15, kb = (L>>4)*8
//   v16bf elements 0..7 = K = kb+0..7, elements 8..15 = K = kb+16..23
// Two contiguous 8xbf16 runs per lane (kb, kb+16) -> two 16B loads.
__device__ __forceinline__ v16bf load_frag_bt(const bf16_t* __restrict__ p, int ld) {
  const int lane = threadIdx.x & 31;
  const int n  = lane & 15;
  const int kb = (lane >> 4) << 3;
  const bf16_t* b = p + n * ld + kb;
  v8bf lo = *(const v8bf*)(b);
  v8bf hi = *(const v8bf*)(b + 16);
  v16bf r;
#pragma unroll
  for (int e = 0; e < 8; ++e) { r[e] = lo[e]; r[e + 8] = hi[e]; }
  return r;
}

__device__ __forceinline__ v8f wmma_bf16(v16bf a, v16bf b, v8f c) {
  return __builtin_amdgcn_wmma_f32_16x16x32_bf16(
      /*neg_a=*/false, a, /*neg_b=*/false, b,
      /*c_mod=*/(short)0, c, /*reuse_a=*/false, /*reuse_b=*/false);
}

// D/C layout (16x16 f32): lane L -> col n = L&15 ; VGPR v -> row (L>>4)*8 + v
__device__ __forceinline__ void store_tile_bf16(bf16_t* __restrict__ C, int ldc,
                                                int m0, int n0, v8f c) {
  const int lane = threadIdx.x & 31;
  const int n = n0 + (lane & 15);
  const int mb = m0 + ((lane >> 4) << 3);
#pragma unroll
  for (int v = 0; v < 8; ++v)
    C[(size_t)(mb + v) * ldc + n] = (bf16_t)c[v];
}

// ---------------------------------------------------------------------------
// JAX threefry2x32-20 dropout mask, key = jax.random.key(123) -> (ks0,ks1)=(0,123)
// bernoulli(key, 0.5, (4096,4096)): counts = iota(16M) split into halves;
// element i<8M uses out0 of pair (i, i+8M); i>=8M uses out1 of (i-8M, i).
// uniform = bitcast((bits>>9)|0x3f800000) - 1 ;  keep <=> uniform < 0.5
//         <=> bits[31] == 0  (exact).
// ---------------------------------------------------------------------------
__device__ __forceinline__ unsigned rotl32(unsigned x, int r) {
  return (x << r) | (x >> (32 - r));
}

__device__ __forceinline__ bool tf_keep(unsigned idx) {
  const unsigned ks0 = 0u, ks1 = 123u;
  const unsigned ks2 = 0x1BD11BDAu ^ ks0 ^ ks1;
  const unsigned h = (unsigned)SEQ_T * (unsigned)SEQ_T / 2u;  // 8388608
  unsigned c0, c1;
  bool first;
  if (idx < h) { c0 = idx;      c1 = idx + h; first = true;  }
  else         { c0 = idx - h;  c1 = idx;     first = false; }
  unsigned x0 = c0 + ks0, x1 = c1 + ks1;
#define TF_ROUND(rc) { x0 += x1; x1 = rotl32(x1, (rc)); x1 ^= x0; }
  TF_ROUND(13) TF_ROUND(15) TF_ROUND(26) TF_ROUND(6)
  x0 += ks1; x1 += ks2 + 1u;
  TF_ROUND(17) TF_ROUND(29) TF_ROUND(16) TF_ROUND(24)
  x0 += ks2; x1 += ks0 + 2u;
  TF_ROUND(13) TF_ROUND(15) TF_ROUND(26) TF_ROUND(6)
  x0 += ks0; x1 += ks1 + 3u;
  TF_ROUND(17) TF_ROUND(29) TF_ROUND(16) TF_ROUND(24)
  x0 += ks1; x1 += ks2 + 4u;
  TF_ROUND(13) TF_ROUND(15) TF_ROUND(26) TF_ROUND(6)
  x0 += ks2; x1 += ks0 + 5u;
#undef TF_ROUND
  unsigned bits = first ? x0 : x1;
  return (bits & 0x80000000u) == 0u;
}

// ---------------------------------------------------------------------------
// Kernel 1: f32 -> bf16 conversion, 8 elements/thread (b128 in/out)
// ---------------------------------------------------------------------------
__global__ __launch_bounds__(256) void cvt_f32_bf16_v8(const float4* __restrict__ in,
                                                       v8bf* __restrict__ out, int n8) {
  int i = blockIdx.x * 256 + threadIdx.x;
  if (i < n8) {
    float4 f0 = in[2 * i];
    float4 f1 = in[2 * i + 1];
    v8bf o;
    o[0] = (bf16_t)f0.x; o[1] = (bf16_t)f0.y; o[2] = (bf16_t)f0.z; o[3] = (bf16_t)f0.w;
    o[4] = (bf16_t)f1.x; o[5] = (bf16_t)f1.y; o[6] = (bf16_t)f1.z; o[7] = (bf16_t)f1.w;
    out[i] = o;
  }
}

// ---------------------------------------------------------------------------
// Kernel 2: C[m,n] = sum_k A[m,k] * B[n,k]   (both row-major, NT GEMM)
// One 32x32 tile per wave32 (2 A-frags x 2 B-frags -> 4 WMMAs per k-step;
// halves fragment traffic per WMMA and gives independent accumulators for
// back-to-back WMMA issue). 8 waves per block.
// ---------------------------------------------------------------------------
__global__ __launch_bounds__(256) void gemm_nt_bf16(const bf16_t* __restrict__ A,
                                                    const bf16_t* __restrict__ B,
                                                    bf16_t* __restrict__ C,
                                                    int N, int K) {
  const int wave  = (blockIdx.x * 256 + threadIdx.x) >> 5;
  const int ntile = N >> 5;                 // 32-wide tiles
  const int mt = wave / ntile;
  const int nt = wave % ntile;
  const bf16_t* A0 = A + (size_t)(mt * 32) * K;
  const bf16_t* A1 = A0 + (size_t)16 * K;
  const bf16_t* B0 = B + (size_t)(nt * 32) * K;
  const bf16_t* B1 = B0 + (size_t)16 * K;
  v8f c00 = {}, c01 = {}, c10 = {}, c11 = {};
  for (int k0 = 0; k0 < K; k0 += 32) {
    v16bf a0 = load_frag_a(A0 + k0, K);
    v16bf a1 = load_frag_a(A1 + k0, K);
    v16bf b0 = load_frag_bt(B0 + k0, K);
    v16bf b1 = load_frag_bt(B1 + k0, K);
    c00 = wmma_bf16(a0, b0, c00);
    c01 = wmma_bf16(a0, b1, c01);
    c10 = wmma_bf16(a1, b0, c10);
    c11 = wmma_bf16(a1, b1, c11);
  }
  store_tile_bf16(C, N, mt * 32,      nt * 32,      c00);
  store_tile_bf16(C, N, mt * 32,      nt * 32 + 16, c01);
  store_tile_bf16(C, N, mt * 32 + 16, nt * 32,      c10);
  store_tile_bf16(C, N, mt * 32 + 16, nt * 32 + 16, c11);
}

// ---------------------------------------------------------------------------
// Kernel 3: fused attention over a 16-row strip.
// LDS: 16x4096 f32 scores (256KB) + 16x1024 bf16 Q strip (32KB) = 288KB,
// exploiting CDNA5's 320KB-per-WGP LDS. 8 waves:
//   - stage Q strip
//   - each wave computes 32 of the 256 16x16 score tiles via WMMA,
//     two j-tiles at a time (A-fragment from LDS reused across two B-frags)
//   - each wave does causal softmax + threefry dropout for 2 rows
// ---------------------------------------------------------------------------
__global__ __launch_bounds__(256) void attn_fused(const bf16_t* __restrict__ Q,
                                                  const bf16_t* __restrict__ Km,
                                                  float* __restrict__ out) {
  extern __shared__ char smem[];
  float*  sc = (float*)smem;                                  // [16][4096]
  bf16_t* qs = (bf16_t*)(smem + 16 * SEQ_T * sizeof(float));  // [16][1024]

  const int tid  = threadIdx.x;
  const int row0 = blockIdx.x * 16;

  // ---- stage Q strip (32KB, 16B vectors) ----
  {
    const uint4* src = (const uint4*)(Q + (size_t)row0 * DIM_D);
    uint4* dst = (uint4*)qs;
    for (int i = tid; i < (16 * DIM_D) / 8; i += 256) dst[i] = src[i];
  }
  __syncthreads();

  const int wave = tid >> 5;
  const int lane = tid & 31;

  // ---- scores: S = (Q Kb^T) / sqrt(d) into LDS; 2 j-tiles per pass ----
  for (int jt = wave * 2; jt < SEQ_T / 16; jt += 16) {
    v8f c0 = {}, c1 = {};
    const bf16_t* B0 = Km + (size_t)(jt * 16) * DIM_D;
    const bf16_t* B1 = B0 + (size_t)16 * DIM_D;
    for (int k0 = 0; k0 < DIM_D; k0 += 32) {
      v16bf a  = load_frag_a(qs + k0, DIM_D);   // ds_load_b128 x2, reused 2x
      v16bf b0 = load_frag_bt(B0 + k0, DIM_D);
      v16bf b1 = load_frag_bt(B1 + k0, DIM_D);
      c0 = wmma_bf16(a, b0, c0);
      c1 = wmma_bf16(a, b1, c1);
    }
    const int n  = jt * 16 + (lane & 15);
    const int mb = (lane >> 4) << 3;
#pragma unroll
    for (int v = 0; v < 8; ++v) {
      sc[(mb + v) * SEQ_T + n]      = c0[v] * 0.03125f;  // 1/sqrt(1024)
      sc[(mb + v) * SEQ_T + n + 16] = c1[v] * 0.03125f;
    }
  }
  __syncthreads();

  // ---- causal softmax + renorm + dropout; wave handles rows 2w, 2w+1 ----
#pragma unroll
  for (int rr = 0; rr < 2; ++rr) {
    const int r  = wave * 2 + rr;
    const int gi = row0 + r;
    const float* srow = sc + (size_t)r * SEQ_T;

    float m = -3.402823466e38f;
    for (int j = lane; j <= gi; j += 32) m = fmaxf(m, srow[j]);
#pragma unroll
    for (int off = 16; off; off >>= 1) m = fmaxf(m, __shfl_xor(m, off, 32));

    float s = 0.f;
    for (int j = lane; j <= gi; j += 32) s += __expf(srow[j] - m);
#pragma unroll
    for (int off = 16; off; off >>= 1) s += __shfl_xor(s, off, 32);

    const float inv = 2.0f / s;  // 1/sum * 1/(1-p)
    float* orow = out + (size_t)gi * SEQ_T;
    for (int j = lane; j < SEQ_T; j += 32) {
      float val = 0.f;
      if (j <= gi) {
        val = __expf(srow[j] - m) * inv;
        if (!tf_keep((unsigned)gi * (unsigned)SEQ_T + (unsigned)j)) val = 0.f;
      }
      orow[j] = val;
    }
  }
}

// ---------------------------------------------------------------------------
// Launch
// ---------------------------------------------------------------------------
extern "C" void kernel_launch(void* const* d_in, const int* in_sizes, int n_in,
                              void* d_out, int out_size, void* d_ws, size_t ws_size,
                              hipStream_t stream) {
  (void)in_sizes; (void)n_in; (void)out_size; (void)ws_size;
  const float* x  = (const float*)d_in[0];
  const float* Wq = (const float*)d_in[1];
  const float* Wk = (const float*)d_in[2];
  // d_in[3] (W_value) feeds only dead code in the reference; unused.
  float* out = (float*)d_out;

  // Workspace layout (bf16): xb[4096*1024] wqb[1M] wkb[1M] qb[4096*1024] kb[4096*1024]
  bf16_t* xb  = (bf16_t*)d_ws;
  bf16_t* wqb = xb  + (size_t)SEQ_T * DIM_D;
  bf16_t* wkb = wqb + (size_t)DIM_D * DIM_D;
  bf16_t* qb  = wkb + (size_t)DIM_D * DIM_D;
  bf16_t* kb  = qb  + (size_t)SEQ_T * DIM_D;

  cvt_f32_bf16_v8<<<(SEQ_T * DIM_D / 8) / 256, 256, 0, stream>>>(
      (const float4*)x, (v8bf*)xb, SEQ_T * DIM_D / 8);
  cvt_f32_bf16_v8<<<(DIM_D * DIM_D / 8) / 256, 256, 0, stream>>>(
      (const float4*)Wq, (v8bf*)wqb, DIM_D * DIM_D / 8);
  cvt_f32_bf16_v8<<<(DIM_D * DIM_D / 8) / 256, 256, 0, stream>>>(
      (const float4*)Wk, (v8bf*)wkb, DIM_D * DIM_D / 8);

  // Q = x Wq^T, K = x Wk^T : 4096x1024 output, 32x32 tiles -> 4096 tiles, 8 waves/block
  const int tiles  = (SEQ_T / 32) * (DIM_D / 32);
  const int blocks = tiles / 8;
  gemm_nt_bf16<<<blocks, 256, 0, stream>>>(xb, wqb, qb, DIM_D, DIM_D);
  gemm_nt_bf16<<<blocks, 256, 0, stream>>>(xb, wkb, kb, DIM_D, DIM_D);

  // Fused scores + softmax + causal renorm + dropout. 288KB dynamic LDS.
  const size_t lds_bytes = (size_t)16 * SEQ_T * sizeof(float) +
                           (size_t)16 * DIM_D * sizeof(bf16_t);
  attn_fused<<<SEQ_T / 16, 256, lds_bytes, stream>>>(qb, kb, out);
}
